// PointnetSAModulePairwise_4209067950524
// MI455X (gfx1250) — matrix operand order repaired and checked by reading, not compiled, not run on hardware
//
#include <hip/hip_runtime.h>

// ---------------- problem constants (match reference) ----------------
#define RADIUS   0.4f
#define NSAMPLE  32
#define NPOINT   256
#define SPLIT    18
#define BN_EPS   1e-5f
#define BATCH    2
#define NPTS     8192
#define NFEAT    128
#define CIN      259                      // 3 + 128 + 128
#define KPAD     288                      // CIN padded to multiple of 32
#define NCOL     (BATCH * NPOINT * NSAMPLE)   // 16384 GEMM columns

// ---------------- d_out element offsets (flat, return order) ----------------
// new_xyz (B,256,3) | new_features (B,18,64,256,32) | idx (B,256,32) | grouped (B,259,256,32)
#define OFF_NEWXYZ   0
#define OFF_NEWFEAT  1536
#define OFF_IDX      (OFF_NEWFEAT + (size_t)BATCH * SPLIT * 64 * NPOINT * NSAMPLE) // 18875904
#define OFF_GROUPED  (OFF_IDX + (size_t)BATCH * NPOINT * NSAMPLE)                  // 18892288

typedef __attribute__((ext_vector_type(16))) _Float16 v16h;
typedef __attribute__((ext_vector_type(8)))  _Float16 v8h;
typedef __attribute__((ext_vector_type(8)))  float    v8f;

// =====================================================================
// 1) Farthest point sampling: one 1024-thread block per batch.
//    All 8192 points live in registers (8 per thread); LDS argmax tree.
// =====================================================================
__global__ void fps_kernel(const float* __restrict__ xyz,
                           int* __restrict__ inds,
                           float* __restrict__ newxyz)
{
    const int b = blockIdx.x;
    const int t = threadIdx.x;                 // 0..1023
    __shared__ float sval[1024];
    __shared__ int   sidx[1024];
    __shared__ float bc[3];

    const float* base = xyz + (size_t)b * NPTS * 3;
    float px[8], py[8], pz[8], dist[8];
#pragma unroll
    for (int k = 0; k < 8; ++k) {
        const int p = k * 1024 + t;
        px[k] = base[p * 3 + 0];
        py[k] = base[p * 3 + 1];
        pz[k] = base[p * 3 + 2];
        dist[k] = 1e10f;
    }
    if (t == 0) {
        bc[0] = base[0]; bc[1] = base[1]; bc[2] = base[2];
        inds[b * NPOINT] = 0;
        newxyz[((size_t)b * NPOINT) * 3 + 0] = base[0];
        newxyz[((size_t)b * NPOINT) * 3 + 1] = base[1];
        newxyz[((size_t)b * NPOINT) * 3 + 2] = base[2];
    }
    __syncthreads();

    for (int it = 1; it < NPOINT; ++it) {
        const float lx = bc[0], ly = bc[1], lz = bc[2];
        float best = -1.0f; int bi = t;        // per-thread argmax (ascending p => first-max)
#pragma unroll
        for (int k = 0; k < 8; ++k) {
            const float dx = px[k] - lx, dy = py[k] - ly, dz = pz[k] - lz;
            const float d = dx * dx + dy * dy + dz * dz;
            dist[k] = fminf(dist[k], d);
            if (dist[k] > best) { best = dist[k]; bi = k * 1024 + t; }
        }
        sval[t] = best; sidx[t] = bi;
        __syncthreads();
        for (int sft = 512; sft > 0; sft >>= 1) {
            if (t < sft) {
                const float v2 = sval[t + sft]; const int i2 = sidx[t + sft];
                if (v2 > sval[t] || (v2 == sval[t] && i2 < sidx[t])) { sval[t] = v2; sidx[t] = i2; }
            }
            __syncthreads();
        }
        const int win = sidx[0];
        __syncthreads();
        if ((win & 1023) == t) {               // owner broadcasts winning coords
            const int k = win >> 10;
            bc[0] = px[k]; bc[1] = py[k]; bc[2] = pz[k];
        }
        if (t == 0) inds[b * NPOINT + it] = win;
        __syncthreads();
        if (t == 0) {
            newxyz[((size_t)b * NPOINT + it) * 3 + 0] = bc[0];
            newxyz[((size_t)b * NPOINT + it) * 3 + 1] = bc[1];
            newxyz[((size_t)b * NPOINT + it) * 3 + 2] = bc[2];
        }
        __syncthreads();
    }
}

// =====================================================================
// 2) Ball query: one wave32 per query point; ballot + prefix popcount
//    gives the ordered "first NSAMPLE inside radius" semantics.
// =====================================================================
__global__ void ballquery_kernel(const float* __restrict__ xyz,
                                 const float* __restrict__ newxyz,
                                 int* __restrict__ idx)
{
    const int qp = blockIdx.x;                 // 0 .. B*NPOINT-1
    const int b = qp / NPOINT;
    const int lane = threadIdx.x;              // 0..31 (one wave)
    const float* q = newxyz + (size_t)qp * 3;
    const float qx = q[0], qy = q[1], qz = q[2];
    const float* base = xyz + (size_t)b * NPTS * 3;
    __shared__ int found[NSAMPLE];

    int count = 0;
    for (int off = 0; off < NPTS && count < NSAMPLE; off += 32) {
        const int p = off + lane;
        const float dx = base[p * 3 + 0] - qx;
        const float dy = base[p * 3 + 1] - qy;
        const float dz = base[p * 3 + 2] - qz;
        const bool inside = (dx * dx + dy * dy + dz * dz) < (RADIUS * RADIUS);
        const unsigned mask = (unsigned)__ballot(inside);   // wave32: low 32 bits
        const int pre = __popc(mask & ((1u << lane) - 1u));
        if (inside && (count + pre) < NSAMPLE) found[count + pre] = p;
        count += __popc(mask);
    }
    __syncthreads();
    const int total = count < NSAMPLE ? count : NSAMPLE;
    const int first = (total > 0) ? found[0] : 0;
    idx[(size_t)qp * NSAMPLE + lane] = (lane < total) ? found[lane] : first;
}

// =====================================================================
// 3) Grouping: build grouped (d_out) AND the f16 GEMM activation matrix
//    X1 stored column-major-in-K: X1[col][KPAD], col = ((b*P)+p)*S + smp.
// =====================================================================
__global__ void group_kernel(const float* __restrict__ xyz,
                             const float* __restrict__ feat,
                             const int* __restrict__ idx,
                             const int* __restrict__ inds,
                             const float* __restrict__ newxyz,
                             float* __restrict__ grouped,
                             _Float16* __restrict__ X1)
{
    const int col = blockIdx.x * 256 + threadIdx.x;   // 0..16383
    const int b = col >> 13;
    const int p = (col >> 5) & (NPOINT - 1);
    const int smp = col & (NSAMPLE - 1);

    const int ni = idx[((size_t)b * NPOINT + p) * NSAMPLE + smp];
    const int ci = inds[b * NPOINT + p];
    const float* q = newxyz + ((size_t)b * NPOINT + p) * 3;
    const float* pt = xyz + ((size_t)b * NPTS + ni) * 3;
    float r[3] = { pt[0] - q[0], pt[1] - q[1], pt[2] - q[2] };

    _Float16* xc = X1 + (size_t)col * KPAD;
#pragma unroll
    for (int c = 0; c < 3; ++c) {
        grouped[(((size_t)b * CIN + c) * NPOINT + p) * NSAMPLE + smp] = r[c];
        xc[c] = (_Float16)r[c];
    }
    for (int c = 0; c < NFEAT; ++c) {
        const float fn = feat[((size_t)b * NFEAT + c) * NPTS + ni];
        const float fc = feat[((size_t)b * NFEAT + c) * NPTS + ci];
        grouped[(((size_t)b * CIN + 3 + c) * NPOINT + p) * NSAMPLE + smp] = fn;
        grouped[(((size_t)b * CIN + 131 + c) * NPOINT + p) * NSAMPLE + smp] = fc;
        xc[3 + c]   = (_Float16)fn;
        xc[131 + c] = (_Float16)fc;
    }
    for (int c = CIN; c < KPAD; ++c) xc[c] = (_Float16)0.0f;   // K padding
}

// =====================================================================
// 4) Weight conversion f32 -> f16 (K-padded for layer 1).
// =====================================================================
__global__ void convw_kernel(const float* __restrict__ W1, const float* __restrict__ W2,
                             const float* __restrict__ W3, _Float16* __restrict__ W1h,
                             _Float16* __restrict__ W2h, _Float16* __restrict__ W3h)
{
    const int r = blockIdx.x * 256 + threadIdx.x;     // row = s*64 + o
    if (r >= SPLIT * 64) return;
    const float* w1 = W1 + (size_t)r * CIN;
    _Float16* o1 = W1h + (size_t)r * KPAD;
    for (int c = 0; c < CIN; ++c)  o1[c] = (_Float16)w1[c];
    for (int c = CIN; c < KPAD; ++c) o1[c] = (_Float16)0.0f;
    for (int c = 0; c < 64; ++c) {
        W2h[(size_t)r * 64 + c] = (_Float16)W2[(size_t)r * 64 + c];
        W3h[(size_t)r * 64 + c] = (_Float16)W3[(size_t)r * 64 + c];
    }
}

// =====================================================================
// 5) WMMA GEMM: per wave a 64x16 tile of Y[s] = W[s] (64xLDK) * X (LDKxNCOL).
//    A fragment: one contiguous v16h load (K = 16h..16h+15, same order as B).
//    B fragment: one contiguous v16h load (X is column-major in K).
//    Y layout [s][o][col] so BN stats reduce over contiguous memory.
// =====================================================================
template <int KCHUNKS, int LDK>
__global__ void gemm_wmma_kernel(const _Float16* __restrict__ W,
                                 const _Float16* __restrict__ X,
                                 float* __restrict__ Y,
                                 size_t sstrideX)
{
    const int lane  = threadIdx.x & 31;
    const int wave  = threadIdx.x >> 5;               // 8 waves/block
    const int s     = blockIdx.y;
    const int ncol0 = (blockIdx.x * 8 + wave) * 16;
    const int nlo   = lane & 15;
    const int khalf = lane >> 4;                      // 0: K 0-15 half, 1: K 16-31 half

    const _Float16* Xs   = X + (size_t)s * sstrideX;
    const _Float16* xcol = Xs + (size_t)(ncol0 + nlo) * LDK + khalf * 16;

    v8f acc[4] = {v8f{}, v8f{}, v8f{}, v8f{}};

#pragma unroll
    for (int kc = 0; kc < KCHUNKS; ++kc) {
        const v16h bt = *(const v16h*)(xcol + kc * 32);   // B 32x16 fragment
#pragma unroll
        for (int m = 0; m < 4; ++m) {
            const _Float16* wr = W + ((size_t)s * 64 + m * 16 + nlo) * LDK
                                   + kc * 32 + khalf * 16;
            const v16h at = *(const v16h*)(wr);           // K = 16h..16h+15
            acc[m] = __builtin_amdgcn_wmma_f32_16x16x32_f16(
                false, at, false, bt, (short)0, acc[m], false, false);
        }
    }

    float* y = Y + (size_t)s * 64 * NCOL + ncol0 + nlo;
#pragma unroll
    for (int m = 0; m < 4; ++m) {
#pragma unroll
        for (int v = 0; v < 8; ++v) {
            const int row = m * 16 + v + (khalf ? 8 : 0);  // C/D layout: VGPR v -> M
            y[(size_t)row * NCOL] = acc[m][v];
        }
    }
}

// =====================================================================
// 6) BN stats: one block per (s,o); reduce 16384 contiguous floats.
// =====================================================================
__global__ void bn_stats_kernel(const float* __restrict__ Y, float* __restrict__ stats)
{
    const int so = blockIdx.x;                         // 0 .. SPLIT*64-1
    const float* y = Y + (size_t)so * NCOL;
    float s = 0.f, s2 = 0.f;
    for (int i = threadIdx.x; i < NCOL; i += 256) { const float v = y[i]; s += v; s2 += v * v; }
    __shared__ float sa[256], sb[256];
    sa[threadIdx.x] = s; sb[threadIdx.x] = s2;
    __syncthreads();
    for (int sft = 128; sft > 0; sft >>= 1) {
        if (threadIdx.x < sft) { sa[threadIdx.x] += sa[threadIdx.x + sft]; sb[threadIdx.x] += sb[threadIdx.x + sft]; }
        __syncthreads();
    }
    if (threadIdx.x == 0) {
        const float mean = sa[0] * (1.0f / NCOL);
        const float var  = sb[0] * (1.0f / NCOL) - mean * mean;
        stats[so * 2 + 0] = mean;
        stats[so * 2 + 1] = rsqrtf(var + BN_EPS);
    }
}

// =====================================================================
// 7a) BN-apply + ReLU -> f16 activations X2[s][col][64] for the next GEMM.
// =====================================================================
__global__ void bn_apply_h_kernel(const float* __restrict__ Y, const float* __restrict__ stats,
                                  const float* __restrict__ g, const float* __restrict__ bta,
                                  _Float16* __restrict__ X2)
{
    const size_t tid = (size_t)blockIdx.x * 256 + threadIdx.x;  // (s*64+o)*NCOL + col
    const int col = (int)(tid & (NCOL - 1));
    const int so  = (int)(tid >> 14);
    const int o   = so & 63;
    const int s   = so >> 6;
    float v = Y[tid];
    v = (v - stats[so * 2]) * stats[so * 2 + 1] * g[so] + bta[so];
    v = fmaxf(v, 0.0f);
    X2[((size_t)s * NCOL + col) * 64 + o] = (_Float16)v;
}

// 7b) Final BN-apply + ReLU -> d_out new_features (B,18,64,256,32) f32.
__global__ void bn_apply_out_kernel(const float* __restrict__ Y, const float* __restrict__ stats,
                                    const float* __restrict__ g, const float* __restrict__ bta,
                                    float* __restrict__ out)
{
    const size_t tid = (size_t)blockIdx.x * 256 + threadIdx.x;
    const int col = (int)(tid & (NCOL - 1));
    const int so  = (int)(tid >> 14);
    const int o   = so & 63;
    const int s   = so >> 6;
    float v = Y[tid];
    v = (v - stats[so * 2]) * stats[so * 2 + 1] * g[so] + bta[so];
    v = fmaxf(v, 0.0f);
    const int b = col >> 13;                    // col = (b*256+p)*32+smp ; 8192 per batch
    const int rem = col & 8191;
    out[(((size_t)b * SPLIT + s) * 64 + o) * 8192 + rem] = v;
}

// =====================================================================
// host: workspace layout + launch sequence
// =====================================================================
extern "C" void kernel_launch(void* const* d_in, const int* in_sizes, int n_in,
                              void* d_out, int out_size, void* d_ws, size_t ws_size,
                              hipStream_t stream)
{
    (void)in_sizes; (void)n_in; (void)out_size;
    const float* xyz  = (const float*)d_in[0];
    const float* feat = (const float*)d_in[1];
    const float* W1 = (const float*)d_in[2];
    const float* g1 = (const float*)d_in[3];
    const float* b1 = (const float*)d_in[4];
    const float* W2 = (const float*)d_in[5];
    const float* g2 = (const float*)d_in[6];
    const float* b2 = (const float*)d_in[7];
    const float* W3 = (const float*)d_in[8];
    const float* g3 = (const float*)d_in[9];
    const float* b3 = (const float*)d_in[10];

    float* outf = (float*)d_out;
    float* out_newxyz  = outf + OFF_NEWXYZ;
    float* out_newfeat = outf + OFF_NEWFEAT;
    int*   out_idx     = (int*)(outf + OFF_IDX);
    float* out_grouped = outf + OFF_GROUPED;

    // ---- workspace layout (256B aligned slabs) ----
    char* ws = (char*)d_ws;
    size_t off = 0;
    auto alloc = [&](size_t bytes) { char* p = ws + off; off += (bytes + 255) & ~(size_t)255; return p; };
    int*      ws_inds  = (int*)     alloc((size_t)BATCH * NPOINT * 4);
    float*    ws_stats = (float*)   alloc((size_t)SPLIT * 64 * 2 * 4);
    _Float16* ws_W1h   = (_Float16*)alloc((size_t)SPLIT * 64 * KPAD * 2);
    _Float16* ws_W2h   = (_Float16*)alloc((size_t)SPLIT * 64 * 64 * 2);
    _Float16* ws_W3h   = (_Float16*)alloc((size_t)SPLIT * 64 * 64 * 2);
    _Float16* ws_X1    = (_Float16*)alloc((size_t)NCOL * KPAD * 2);
    _Float16* ws_X2    = (_Float16*)alloc((size_t)SPLIT * NCOL * 64 * 2);
    float*    ws_Y     = (float*)   alloc((size_t)SPLIT * 64 * NCOL * 4);
    if (off > ws_size) return;   // workspace too small: deterministic no-op

    // geometry + grouping
    convw_kernel<<<(SPLIT * 64 + 255) / 256, 256, 0, stream>>>(W1, W2, W3, ws_W1h, ws_W2h, ws_W3h);
    fps_kernel<<<BATCH, 1024, 0, stream>>>(xyz, ws_inds, out_newxyz);
    ballquery_kernel<<<BATCH * NPOINT, 32, 0, stream>>>(xyz, out_newxyz, out_idx);
    group_kernel<<<NCOL / 256, 256, 0, stream>>>(xyz, feat, out_idx, ws_inds, out_newxyz,
                                                 out_grouped, ws_X1);

    const dim3 ggrid(NCOL / 128, SPLIT);           // 8 waves/block, 16 cols/wave
    const int  napply = SPLIT * 64 * NCOL / 256;   // 73728 blocks

    // layer 1: K = 288 (9 chunks), X shared across splits
    gemm_wmma_kernel<9, KPAD><<<ggrid, 256, 0, stream>>>(ws_W1h, ws_X1, ws_Y, (size_t)0);
    bn_stats_kernel<<<SPLIT * 64, 256, 0, stream>>>(ws_Y, ws_stats);
    bn_apply_h_kernel<<<napply, 256, 0, stream>>>(ws_Y, ws_stats, g1, b1, ws_X2);

    // layer 2: K = 64 (2 chunks), per-split X
    gemm_wmma_kernel<2, 64><<<ggrid, 256, 0, stream>>>(ws_W2h, ws_X2, ws_Y, (size_t)NCOL * 64);
    bn_stats_kernel<<<SPLIT * 64, 256, 0, stream>>>(ws_Y, ws_stats);
    bn_apply_h_kernel<<<napply, 256, 0, stream>>>(ws_Y, ws_stats, g2, b2, ws_X2);

    // layer 3: K = 64, final output straight to d_out
    gemm_wmma_kernel<2, 64><<<ggrid, 256, 0, stream>>>(ws_W3h, ws_X2, ws_Y, (size_t)NCOL * 64);
    bn_stats_kernel<<<SPLIT * 64, 256, 0, stream>>>(ws_Y, ws_stats);
    bn_apply_out_kernel<<<napply, 256, 0, stream>>>(ws_Y, ws_stats, g3, b3, out_newfeat);
}
